// Decoder_87205015978746
// MI455X (gfx1250) — matrix operand order repaired and checked by reading, not compiled, not run on hardware
//
#include <hip/hip_runtime.h>
#include <hip/hip_bf16.h>

// ---------------------------------------------------------------------------
// GRU decoder on MI455X (gfx1250, wave32, WMMA).
//   h0 = tanh(z @ W_z2h^T + b_z2h)
//   xp[t,b,:] = x[b,t,:] @ W_ih^T + b_ih          (big parallel GEMM, f16 ws)
//   for t: hp = h @ W_hh^T + b_hh; gates; h'      (512 serial WMMA steps)
//   y[b,t,:] = sigmoid(hs[t,b,:] @ W_out^T + b_out)
// All GEMMs: v_wmma_f32_16x16x32_f16, f32 accumulation, double-buffered LDS
// with software-pipelined async (ASYNCcnt) global->LDS staging.
// ---------------------------------------------------------------------------

typedef _Float16 h8v  __attribute__((ext_vector_type(8)));
typedef _Float16 v16h __attribute__((ext_vector_type(16)));
typedef float    v8f  __attribute__((ext_vector_type(8)));
typedef int      gv2i __attribute__((vector_size(8)));    // matches builtin param
typedef int      gv4i __attribute__((vector_size(16)));   // matches builtin param

#define B_  256
#define T_  512
#define I_  128
#define H_  1024
#define L_  256
#define G_  (3*H_)   // 3072 gate columns

#define AS1 __attribute__((address_space(1)))
#define AS3 __attribute__((address_space(3)))

// ---- gfx1250 async global->LDS copies (ASYNCcnt path), with safe fallback --
#if defined(__has_builtin)
#  if __has_builtin(__builtin_amdgcn_global_load_async_to_lds_b64) && \
      __has_builtin(__builtin_amdgcn_global_load_async_to_lds_b128)
#    define USE_ASYNC_LDS 1
#  endif
#  if __has_builtin(__builtin_amdgcn_s_wait_asynccnt)
#    define HAVE_WAIT_ASYNC_BUILTIN 1
#  endif
#endif
#ifndef USE_ASYNC_LDS
#  define USE_ASYNC_LDS 0
#endif

#if USE_ASYNC_LDS
__device__ __forceinline__ void async_cp_b64(_Float16* ldst, const _Float16* gsrc) {
  __builtin_amdgcn_global_load_async_to_lds_b64((AS1 gv2i*)(void*)gsrc,
                                                (AS3 gv2i*)(void*)ldst, 0, 0);
}
__device__ __forceinline__ void async_cp_b128(_Float16* ldst, const _Float16* gsrc) {
  __builtin_amdgcn_global_load_async_to_lds_b128((AS1 gv4i*)(void*)gsrc,
                                                 (AS3 gv4i*)(void*)ldst, 0, 0);
}
__device__ __forceinline__ void wait_async0() {
#ifdef HAVE_WAIT_ASYNC_BUILTIN
  __builtin_amdgcn_s_wait_asynccnt(0);
#else
  asm volatile("s_wait_asynccnt 0x0" ::: "memory");
#endif
}
#endif

__device__ __forceinline__ float sigmoidf_(float x) { return 1.0f / (1.0f + __expf(-x)); }

// --- WMMA fragment loads from LDS tiles (K-chunk = 32, f16) -----------------
// A tile LDS layout: [row][k], 32 halves/row. gfx1250 16-bit A 16x32:
// lane l -> row = l&15, kh = l>>4; halves = K[kh*8..+7] ++ K[16+kh*8..+7].
__device__ __forceinline__ v16h load_fragA(const _Float16* sA, int lane) {
  const int r = lane & 15, kh = lane >> 4;
  h8v lo = *(const h8v*)(sA + r * 32 + kh * 8);
  h8v hi = *(const h8v*)(sA + r * 32 + 16 + kh * 8);
  return __builtin_shufflevector(lo, hi, 0,1,2,3,4,5,6,7,8,9,10,11,12,13,14,15);
}
// B tile LDS layout: [col n][k]. gfx1250 16-bit B 32x16:
// lane l -> n = l&15, kh = l>>4; halves = K[kh*8..+7] ++ K[16+kh*8..+7].
__device__ __forceinline__ v16h load_fragB(const _Float16* sB, int lane) {
  const int n = lane & 15, kh = lane >> 4;
  h8v lo = *(const h8v*)(sB + n * 32 + kh * 8);
  h8v hi = *(const h8v*)(sB + n * 32 + 16 + kh * 8);
  return __builtin_shufflevector(lo, hi, 0,1,2,3,4,5,6,7,8,9,10,11,12,13,14,15);
}
__device__ __forceinline__ v8f wmma_f16(v16h a, v16h b, v8f c) {
  return __builtin_amdgcn_wmma_f32_16x16x32_f16(false, a, false, b, (short)0, c,
                                                false, false);
}

// --- Generic block GEMM: 256 threads, 8 waves (2x4 of 16x16 tiles) ----------
// C tile = 32x64 of A(MxK) * B^T; B stored n-major K-contiguous (native
// layout of all weights). Double-buffered LDS, pipelined staging.
template <int K>
__device__ __forceinline__ v8f gemm_acc_32x64(const _Float16* __restrict__ A,
                                              const _Float16* __restrict__ Bm,
                                              long m0, long n0,
                                              _Float16* sA /*2*32*32*/,
                                              _Float16* sB /*2*64*32*/) {
  const int tid = threadIdx.x, lane = tid & 31, w = tid >> 5;
  const int wr = (w >> 2) * 16, wc = (w & 3) * 16;
  const _Float16* gA = A + (m0 + (tid >> 3)) * (long)K + (tid & 7) * 4;   // 4 halves
  const _Float16* gB = Bm + (n0 + (tid >> 2)) * (long)K + (tid & 3) * 8;  // 8 halves
  const int aoff = (tid >> 3) * 32 + (tid & 7) * 4;
  const int boff = (tid >> 2) * 32 + (tid & 3) * 8;
  v8f c = {};
  int buf = 0;
#if USE_ASYNC_LDS
  async_cp_b64(sA + aoff, gA);
  async_cp_b128(sB + boff, gB);
  wait_async0();
  __syncthreads();
  for (int kc = 0; kc < K; kc += 32) {
    const bool more = (kc + 32 < K);
    if (more) {
      async_cp_b64(sA + (buf ^ 1) * (32 * 32) + aoff, gA + kc + 32);
      async_cp_b128(sB + (buf ^ 1) * (64 * 32) + boff, gB + kc + 32);
    }
    v16h a = load_fragA(sA + buf * (32 * 32) + wr * 32, lane);
    v16h b = load_fragB(sB + buf * (64 * 32) + wc * 32, lane);
    c = wmma_f16(a, b, c);
    if (more) { wait_async0(); __syncthreads(); buf ^= 1; }
  }
#else
  uint2 ra = *(const uint2*)gA;
  uint4 rb = *(const uint4*)gB;
  *(uint2*)(sA + aoff) = ra;
  *(uint4*)(sB + boff) = rb;
  __syncthreads();
  for (int kc = 0; kc < K; kc += 32) {
    const bool more = (kc + 32 < K);
    if (more) { ra = *(const uint2*)(gA + kc + 32); rb = *(const uint4*)(gB + kc + 32); }
    v16h a = load_fragA(sA + buf * (32 * 32) + wr * 32, lane);
    v16h b = load_fragB(sB + buf * (64 * 32) + wc * 32, lane);
    c = wmma_f16(a, b, c);
    if (more) {
      *(uint2*)(sA + (buf ^ 1) * (32 * 32) + aoff) = ra;
      *(uint4*)(sB + (buf ^ 1) * (64 * 32) + boff) = rb;
      __syncthreads();
      buf ^= 1;
    }
  }
#endif
  return c;
}

// --- elementwise conversion / repack ---------------------------------------
__global__ void f32_to_f16_kernel(const float* __restrict__ src,
                                  _Float16* __restrict__ dst, long n) {
  long i = (long)blockIdx.x * blockDim.x + threadIdx.x;
  if (i < n) dst[i] = (_Float16)src[i];
}

// xh[(t*B + b)*I + i] = target[(b*T + t)*I + i]
__global__ void pack_x_kernel(const float* __restrict__ tgt,
                              _Float16* __restrict__ xh) {
  const int t = blockIdx.x, b = blockIdx.y, i = threadIdx.x;
  xh[((size_t)t * B_ + b) * I_ + i] = (_Float16)tgt[((size_t)b * T_ + t) * I_ + i];
}

// --- h0 = tanh(z @ W_z2h^T + b_z2h) ----------------------------------------
__global__ __launch_bounds__(256) void h0_kernel(const _Float16* __restrict__ zh,
                                                 const _Float16* __restrict__ Wz,
                                                 const float* __restrict__ bz,
                                                 _Float16* __restrict__ h0) {
  __shared__ __align__(64) _Float16 sA[2 * 32 * 32];
  __shared__ __align__(64) _Float16 sB[2 * 64 * 32];
  const long m0 = (long)blockIdx.y * 32, n0 = (long)blockIdx.x * 64;
  v8f c = gemm_acc_32x64<L_>(zh, Wz, m0, n0, sA, sB);
  const int lane = threadIdx.x & 31, w = threadIdx.x >> 5;
  const int wr = (w >> 2) * 16, wc = (w & 3) * 16;
  const int nl = lane & 15, mh = lane >> 4;
#pragma unroll
  for (int i = 0; i < 8; ++i) {
    const long m = m0 + wr + mh * 8 + i;
    const long j = n0 + wc + nl;
    h0[m * H_ + j] = (_Float16)tanhf(c[i] + bz[j]);
  }
}

// --- xp[t*B+b, g] = x @ W_ih^T + b_ih  (write-once stream: NT stores) ------
__global__ __launch_bounds__(256) void xproj_kernel(const _Float16* __restrict__ xh,
                                                    const _Float16* __restrict__ Wih,
                                                    const float* __restrict__ bih,
                                                    _Float16* __restrict__ xp) {
  __shared__ __align__(64) _Float16 sA[2 * 32 * 32];
  __shared__ __align__(64) _Float16 sB[2 * 64 * 32];
  const long m0 = (long)blockIdx.y * 32, n0 = (long)blockIdx.x * 64;
  v8f c = gemm_acc_32x64<I_>(xh, Wih, m0, n0, sA, sB);
  const int lane = threadIdx.x & 31, w = threadIdx.x >> 5;
  const int wr = (w >> 2) * 16, wc = (w & 3) * 16;
  const int nl = lane & 15, mh = lane >> 4;
#pragma unroll
  for (int i = 0; i < 8; ++i) {
    const long m = m0 + wr + mh * 8 + i;
    const long g = n0 + wc + nl;
    __builtin_nontemporal_store((_Float16)(c[i] + bih[g]), xp + (size_t)m * G_ + g);
  }
}

// --- one GRU step: hp = h @ W_hh^T + b_hh; gates; h', hs[t] ----------------
// Block owns 32 batch rows x 64 h cols; r/z/n slabs computed together
// (3 WMMA accumulators/wave) so the gate math fuses locally.
__global__ __launch_bounds__(256) void gru_step_kernel(
    const _Float16* __restrict__ hprev, _Float16* __restrict__ hnext,
    const _Float16* __restrict__ Whh, const _Float16* __restrict__ xp,
    _Float16* __restrict__ hs, const float* __restrict__ bhh, int t) {
  __shared__ __align__(64) _Float16 sA[2 * 32 * 32];
  __shared__ __align__(64) _Float16 sB[3][2 * 64 * 32];
  const int tid = threadIdx.x, lane = tid & 31, w = tid >> 5;
  const int wr = (w >> 2) * 16, wc = (w & 3) * 16;
  const long m0 = (long)blockIdx.y * 32;   // batch rows
  const long j0 = (long)blockIdx.x * 64;   // h columns
  const _Float16* gA = hprev + (m0 + (tid >> 3)) * H_ + (tid & 7) * 4;
  const long bcol = j0 + (tid >> 2);
  const _Float16* gB0 = Whh + (bcol) * H_ + (tid & 3) * 8;
  const _Float16* gB1 = gB0 + (long)H_ * H_;
  const _Float16* gB2 = gB1 + (long)H_ * H_;
  const int aoff = (tid >> 3) * 32 + (tid & 7) * 4;
  const int boff = (tid >> 2) * 32 + (tid & 3) * 8;
  v8f cr = {}, cz = {}, cn = {};
  int buf = 0;
#if USE_ASYNC_LDS
  async_cp_b64(sA + aoff, gA);
  async_cp_b128(&sB[0][boff], gB0);
  async_cp_b128(&sB[1][boff], gB1);
  async_cp_b128(&sB[2][boff], gB2);
  wait_async0();
  __syncthreads();
  for (int kc = 0; kc < H_; kc += 32) {
    const bool more = (kc + 32 < H_);
    if (more) {
      const int nb = buf ^ 1;
      async_cp_b64(sA + nb * (32 * 32) + aoff, gA + kc + 32);
      async_cp_b128(&sB[0][nb * (64 * 32) + boff], gB0 + kc + 32);
      async_cp_b128(&sB[1][nb * (64 * 32) + boff], gB1 + kc + 32);
      async_cp_b128(&sB[2][nb * (64 * 32) + boff], gB2 + kc + 32);
    }
    v16h a  = load_fragA(sA + buf * (32 * 32) + wr * 32, lane);
    v16h br = load_fragB(&sB[0][buf * (64 * 32) + wc * 32], lane);
    v16h bz = load_fragB(&sB[1][buf * (64 * 32) + wc * 32], lane);
    v16h bn = load_fragB(&sB[2][buf * (64 * 32) + wc * 32], lane);
    cr = wmma_f16(a, br, cr);
    cz = wmma_f16(a, bz, cz);
    cn = wmma_f16(a, bn, cn);
    if (more) { wait_async0(); __syncthreads(); buf ^= 1; }
  }
#else
  uint2 ra = *(const uint2*)gA;
  uint4 rb0 = *(const uint4*)gB0;
  uint4 rb1 = *(const uint4*)gB1;
  uint4 rb2 = *(const uint4*)gB2;
  *(uint2*)(sA + aoff) = ra;
  *(uint4*)(&sB[0][boff]) = rb0;
  *(uint4*)(&sB[1][boff]) = rb1;
  *(uint4*)(&sB[2][boff]) = rb2;
  __syncthreads();
  for (int kc = 0; kc < H_; kc += 32) {
    const bool more = (kc + 32 < H_);
    if (more) {
      ra  = *(const uint2*)(gA + kc + 32);
      rb0 = *(const uint4*)(gB0 + kc + 32);
      rb1 = *(const uint4*)(gB1 + kc + 32);
      rb2 = *(const uint4*)(gB2 + kc + 32);
    }
    v16h a  = load_fragA(sA + buf * (32 * 32) + wr * 32, lane);
    v16h br = load_fragB(&sB[0][buf * (64 * 32) + wc * 32], lane);
    v16h bz = load_fragB(&sB[1][buf * (64 * 32) + wc * 32], lane);
    v16h bn = load_fragB(&sB[2][buf * (64 * 32) + wc * 32], lane);
    cr = wmma_f16(a, br, cr);
    cz = wmma_f16(a, bz, cz);
    cn = wmma_f16(a, bn, cn);
    if (more) {
      const int nb = buf ^ 1;
      *(uint2*)(sA + nb * (32 * 32) + aoff) = ra;
      *(uint4*)(&sB[0][nb * (64 * 32) + boff]) = rb0;
      *(uint4*)(&sB[1][nb * (64 * 32) + boff]) = rb1;
      *(uint4*)(&sB[2][nb * (64 * 32) + boff]) = rb2;
      __syncthreads();
      buf ^= 1;
    }
  }
#endif
  const int nl = lane & 15, mh = lane >> 4;
#pragma unroll
  for (int i = 0; i < 8; ++i) {
    const long m = m0 + wr + mh * 8 + i;   // batch index
    const long j = j0 + wc + nl;           // hidden column
    const size_t xb = ((size_t)t * B_ + m) * G_ + j;
    const float xr = (float)__builtin_nontemporal_load(xp + xb);
    const float xz = (float)__builtin_nontemporal_load(xp + xb + H_);
    const float xn = (float)__builtin_nontemporal_load(xp + xb + 2 * H_);
    const float hr = cr[i] + bhh[j];
    const float hz = cz[i] + bhh[H_ + j];
    const float hn = cn[i] + bhh[2 * H_ + j];
    const float r = sigmoidf_(xr + hr);
    const float u = sigmoidf_(xz + hz);
    const float nd = tanhf(xn + r * hn);
    const float hold = (float)hprev[(size_t)m * H_ + j];
    const float hnew = (1.0f - u) * nd + u * hold;
    hnext[(size_t)m * H_ + j] = (_Float16)hnew;
    __builtin_nontemporal_store((_Float16)hnew, hs + ((size_t)t * B_ + m) * H_ + j);
  }
}

// --- y[b,t,i] = sigmoid(hs[t,b,:] @ W_out^T + b_out) -----------------------
__global__ __launch_bounds__(256) void out_kernel(const _Float16* __restrict__ hsb,
                                                  const _Float16* __restrict__ Wout,
                                                  const float* __restrict__ bout,
                                                  float* __restrict__ out) {
  __shared__ __align__(64) _Float16 sA[2 * 32 * 32];
  __shared__ __align__(64) _Float16 sB[2 * 64 * 32];
  const long m0 = (long)blockIdx.y * 32, n0 = (long)blockIdx.x * 64;
  v8f c = gemm_acc_32x64<H_>(hsb, Wout, m0, n0, sA, sB);
  const int lane = threadIdx.x & 31, w = threadIdx.x >> 5;
  const int wr = (w >> 2) * 16, wc = (w & 3) * 16;
  const int nl = lane & 15, mh = lane >> 4;
#pragma unroll
  for (int i = 0; i < 8; ++i) {
    const long m = m0 + wr + mh * 8 + i;   // m = t*B + b
    const long ic = n0 + wc + nl;          // output feature
    const long b = m & (B_ - 1), t = m >> 8;
    out[((size_t)b * T_ + t) * I_ + ic] = sigmoidf_(c[i] + bout[ic]);
  }
}

// ---------------------------------------------------------------------------
extern "C" void kernel_launch(void* const* d_in, const int* in_sizes, int n_in,
                              void* d_out, int out_size, void* d_ws, size_t ws_size,
                              hipStream_t stream) {
  (void)in_sizes; (void)n_in; (void)out_size; (void)ws_size;
  const float* z      = (const float*)d_in[0];
  const float* target = (const float*)d_in[1];
  const float* W_z2h  = (const float*)d_in[2];
  const float* b_z2h  = (const float*)d_in[3];
  const float* W_ih   = (const float*)d_in[4];
  const float* b_ih   = (const float*)d_in[5];
  const float* W_hh   = (const float*)d_in[6];
  const float* b_hh   = (const float*)d_in[7];
  const float* W_out  = (const float*)d_in[8];
  const float* b_out  = (const float*)d_in[9];
  float* out = (float*)d_out;

  // workspace carve-out (f16 units), ~1.12 GB total
  _Float16* p = (_Float16*)d_ws;
  auto take = [&](size_t n) { _Float16* q = p; p += n; return q; };
  _Float16* zh     = take((size_t)B_ * L_);
  _Float16* Wz2h16 = take((size_t)H_ * L_);
  _Float16* Wih16  = take((size_t)G_ * I_);
  _Float16* Whh16  = take((size_t)G_ * H_);
  _Float16* Wout16 = take((size_t)I_ * H_);
  _Float16* hA     = take((size_t)B_ * H_);
  _Float16* hB     = take((size_t)B_ * H_);
  _Float16* xh     = take((size_t)T_ * B_ * I_);
  _Float16* hs     = take((size_t)T_ * B_ * H_);
  _Float16* xp     = take((size_t)T_ * B_ * G_);

  auto cvt = [&](const float* s, _Float16* d, long n) {
    f32_to_f16_kernel<<<dim3((unsigned)((n + 255) / 256)), 256, 0, stream>>>(s, d, n);
  };
  cvt(z, zh, (long)B_ * L_);
  cvt(W_z2h, Wz2h16, (long)H_ * L_);
  cvt(W_ih, Wih16, (long)G_ * I_);
  cvt(W_hh, Whh16, (long)G_ * H_);
  cvt(W_out, Wout16, (long)I_ * H_);
  pack_x_kernel<<<dim3(T_, B_), I_, 0, stream>>>(target, xh);

  h0_kernel<<<dim3(H_ / 64, B_ / 32), 256, 0, stream>>>(zh, Wz2h16, b_z2h, hA);
  xproj_kernel<<<dim3(G_ / 64, (T_ * B_) / 32), 256, 0, stream>>>(xh, Wih16, b_ih, xp);
  for (int t = 0; t < T_; ++t) {
    const _Float16* hin = (t & 1) ? hB : hA;
    _Float16* hout      = (t & 1) ? hA : hB;
    gru_step_kernel<<<dim3(H_ / 64, B_ / 32), 256, 0, stream>>>(hin, hout, Whh16,
                                                                xp, hs, b_hh, t);
  }
  out_kernel<<<dim3(I_ / 64, (T_ * B_) / 32), 256, 0, stream>>>(hs, Wout16, b_out, out);
}
